// MultiHeadAttention_3728031613617
// MI455X (gfx1250) — matrix operand order repaired and checked by reading, not compiled, not run on hardware
//
#include <hip/hip_runtime.h>
#include <hip/hip_bf16.h>

typedef __attribute__((ext_vector_type(16))) __bf16 v16bf;
typedef __attribute__((ext_vector_type(8)))  float  v8f;
typedef __attribute__((ext_vector_type(4)))  unsigned uint32x4;
typedef __attribute__((ext_vector_type(4)))  int      int32x4;
typedef __attribute__((ext_vector_type(8)))  int      int32x8;

union BFVec { v16bf v; unsigned u[8]; unsigned short s[16]; };

#if defined(__has_builtin)
#if __has_builtin(__builtin_amdgcn_tensor_load_to_lds)
#define USE_TDM 1
#endif
#endif

__device__ __forceinline__ unsigned short f2bf(float f) {
    unsigned u = __float_as_uint(f);
    u += 0x7FFFu + ((u >> 16) & 1u);          // round-to-nearest-even
    return (unsigned short)(u >> 16);
}

__device__ __forceinline__ v8f zero8() {
    v8f z;
#pragma unroll
    for (int i = 0; i < 8; ++i) z[i] = 0.0f;
    return z;
}

__device__ __forceinline__ v8f wmma_bf16(v16bf a, v16bf b, v8f c) {
    // v_wmma_f32_16x16x32_bf16: (neg_a, A, neg_b, B, c_mod, C, reuse_a, reuse_b)
    return __builtin_amdgcn_wmma_f32_16x16x32_bf16(false, a, false, b, (short)0, c,
                                                   false, false);
}

// A fragment (16x32 bf16, M x K) from row-major bf16 matrix.
// Layout: lane m=l&15, kh=l>>4; VGPR j<4: K=8*kh+2j,+1 ; j>=4: K=16+8*kh+2(j-4),+1
__device__ __forceinline__ v16bf load_a_frag(const unsigned short* base, int ld,
                                             int k0, int lane) {
    const int mm = lane & 15, kh = lane >> 4;
    const unsigned short* row = base + mm * ld + k0;
    BFVec f;
#pragma unroll
    for (int j = 0; j < 8; ++j) {
        const int k = (j < 4) ? (8 * kh + 2 * j) : (16 + 8 * kh + 2 * (j - 4));
        f.u[j] = *(const unsigned*)(row + k);
    }
    return f.v;
}

// B fragment (32x16 bf16, K x N) from a [N][K]-stored ("transposed") matrix.
// Layout: lane n=l&15, kh=l>>4; VGPR j<4: K=8*kh+2j,+1 ; j>=4: K=16+8*kh+2(j-4),+1
__device__ __forceinline__ v16bf load_b_fragT(const unsigned short* baseT, int ld,
                                              int n0, int k0, int lane) {
    const int nn = lane & 15, kh = lane >> 4;
    const unsigned short* row = baseT + (n0 + nn) * ld + k0;
    BFVec f;
#pragma unroll
    for (int j = 0; j < 8; ++j) {
        const int k = (j < 4) ? (8 * kh + 2 * j) : (16 + 8 * kh + 2 * (j - 4));
        f.u[j] = *(const unsigned*)(row + k);
    }
    return f.v;
}

// ---------------------------------------------------------------------------
// Kernel 1: fused Q/K/V projection.  x[16384,512] f32 -> Q,K,V [16384,64] bf16
// reads x exactly once (the only mandatory 32MB input pass).
// ---------------------------------------------------------------------------
#define LDX 34   // 32 + 2 pad (17 dwords -> conflict-free-ish)
#define LDW 34

__global__ __launch_bounds__(256) void qkv_kernel(const float* __restrict__ x,
                                                  const float* __restrict__ wk,
                                                  unsigned short* __restrict__ qb,
                                                  unsigned short* __restrict__ kb,
                                                  unsigned short* __restrict__ vb) {
    __shared__ unsigned short xt[128 * LDX];       // x tile, bf16
    __shared__ unsigned short wT[3 * 64 * LDW];    // weights transposed [t][n][k]
    const int tid = threadIdx.x, lane = tid & 31, wave = tid >> 5;
    const int row0 = blockIdx.x * 128;

    v8f acc[3][4];
#pragma unroll
    for (int t = 0; t < 3; ++t)
#pragma unroll
        for (int n = 0; n < 4; ++n) acc[t][n] = zero8();

    for (int k0 = 0; k0 < 512; k0 += 32) {
        __syncthreads();
        {   // stage x tile 128x32 f32 -> bf16 (16 contiguous floats / thread)
            const int r = tid >> 1;
            const int cb = (tid & 1) * 16;
            const float* src = x + (size_t)(row0 + r) * 512 + k0 + cb;
            unsigned short* dst = xt + r * LDX + cb;
#pragma unroll
            for (int i = 0; i < 16; ++i) dst[i] = f2bf(src[i]);
            if (k0 + 32 < 512) __builtin_prefetch(src + 32, 0, 0);  // next x tile
        }
        // stage 3 weight slices 32x64 -> transposed bf16
#pragma unroll
        for (int i = 0; i < 24; ++i) {
            const int lin = i * 256 + tid;            // < 6144
            const int t = lin >> 11;
            const int rem = lin & 2047;
            const int kk = rem >> 6;
            const int n = rem & 63;
            const float v = wk[((size_t)t * 512 + (k0 + kk)) * 64 + n];
            wT[(t * 64 + n) * LDW + kk] = f2bf(v);
        }
        __syncthreads();

        const v16bf a = load_a_frag(xt + (wave * 16) * LDX, LDX, 0, lane);
#pragma unroll
        for (int t = 0; t < 3; ++t)
#pragma unroll
            for (int n = 0; n < 4; ++n) {
                const v16bf b = load_b_fragT(wT + t * 64 * LDW, LDW, n * 16, 0, lane);
                acc[t][n] = wmma_bf16(a, b, acc[t][n]);
            }
    }

    const int nn = lane & 15, hh = lane >> 4;
    unsigned short* outs[3] = {qb, kb, vb};
#pragma unroll
    for (int t = 0; t < 3; ++t)
#pragma unroll
        for (int n = 0; n < 4; ++n)
#pragma unroll
            for (int r = 0; r < 8; ++r) {
                const int row = row0 + wave * 16 + r + 8 * hh;
                outs[t][(size_t)row * 64 + n * 16 + nn] = f2bf(acc[t][n][r]);
            }
}

// ---------------------------------------------------------------------------
// Kernel 2: dense fold.  dense[512,512] f32 -> dr[64,512] bf16
// dr[c][o] = sum_h dense[h*64+c][o]   (tile(V,heads)@dense == V@dr)
// ---------------------------------------------------------------------------
__global__ __launch_bounds__(256) void densered_kernel(const float* __restrict__ dense,
                                                       unsigned short* __restrict__ dr) {
    const int idx = blockIdx.x * 256 + threadIdx.x;   // 0..32767
    const int c = idx >> 9, oo = idx & 511;
    float s = 0.0f;
#pragma unroll
    for (int h = 0; h < 8; ++h) s += dense[(size_t)(h * 64 + c) * 512 + oo];
    dr[idx] = f2bf(s);
}

// ---------------------------------------------------------------------------
// Kernel 3: flash attention.  Q,K,V [8][2048,64] bf16 -> O [8][2048,64] bf16
// block = (batch, 128-row q-tile); 8 waves x 16 rows; 64-key tiles.
// K tile staged by the Tensor Data Mover (TDM); its pad_interval/pad_amount
// reproduce the padded 33-dword LDS row stride (LDT=66 bf16 = 132B).
// ---------------------------------------------------------------------------
#define LDT 66   // 64 + 2 pad (33 dwords stride)

#ifdef USE_TDM
__device__ __forceinline__ void tdm_load_tile64(const unsigned short* gsrc,
                                                unsigned lds_off) {
    const unsigned long long ga = (unsigned long long)(uintptr_t)gsrc;
    uint32x4 g0;
    g0[0] = 1u;                                     // count=1, user mode
    g0[1] = lds_off;                                // lds_addr (bytes)
    g0[2] = (unsigned)(ga & 0xFFFFFFFFu);           // global_addr[31:0]
    g0[3] = (unsigned)((ga >> 32) & 0x1FFFFFFu)     // global_addr[56:32]
          | (2u << 30);                             // type=2 ("image")
    // group1: data_size=8B, pad after every 32 dwords by 1 dword
    const unsigned td0 = 16;       // tensor_dim0: 16 x 8B = 128B row
    const unsigned td1 = 1u << 20; // tensor_dim1: plenty of rows
    const unsigned s0  = 16;       // row stride = 16 x 8B (contiguous rows)
    int32x8 g1;
    g1[0] = (int)((3u << 16) | (1u << 20) | (4u << 22) | (0u << 25));
    g1[1] = (int)((td0 & 0xFFFFu) << 16);                       // [63:48]=td0.lo
    g1[2] = (int)((td0 >> 16) | ((td1 & 0xFFFFu) << 16));       // td0.hi, td1.lo
    g1[3] = (int)((td1 >> 16) | (16u << 16));                   // td1.hi, tile_dim0=16
    g1[4] = (int)(64u);                                         // tile_dim1=64, tile_dim2=0
    g1[5] = (int)s0;                                            // stride0[31:0]
    g1[6] = 0;                                                  // stride0[47:32], stride1.lo
    g1[7] = 0;                                                  // stride1.hi
    int32x4 z4 = {0, 0, 0, 0};
#if __clang_major__ >= 23
    int32x8 z8 = {0, 0, 0, 0, 0, 0, 0, 0};
    __builtin_amdgcn_tensor_load_to_lds(g0, g1, z4, z4, z8, 0);
#else
    __builtin_amdgcn_tensor_load_to_lds(g0, g1, z4, z4, 0);
#endif
}
#endif

__global__ __launch_bounds__(256) void attn_kernel(const unsigned short* __restrict__ qb,
                                                   const unsigned short* __restrict__ kb,
                                                   const unsigned short* __restrict__ vb,
                                                   unsigned short* __restrict__ ob) {
    __shared__ __align__(16) unsigned short ktile[64 * LDT];  // [key][feat] (TDM dest)
    __shared__ __align__(16) unsigned short vtile[64 * LDT];  // [col][key]  (V transposed)
    __shared__ __align__(16) unsigned short ptile[8 * 16 * LDT];  // per-wave P scratch

    const int tid = threadIdx.x, lane = tid & 31, wave = tid >> 5;
    const int b = blockIdx.x >> 4, qt = blockIdx.x & 15;
    const int bbase = b * 2048;
    const int m0 = bbase + qt * 128 + wave * 16;      // global q row
    const int nn = lane & 15, hh = lane >> 4;

    const v16bf qa0 = load_a_frag(qb + (size_t)m0 * 64, 64, 0, lane);
    const v16bf qa1 = load_a_frag(qb + (size_t)m0 * 64, 64, 32, lane);

    v8f o[4];
#pragma unroll
    for (int n = 0; n < 4; ++n) o[n] = zero8();
    float mrow[8], lrow[8];
#pragma unroll
    for (int r = 0; r < 8; ++r) { mrow[r] = -INFINITY; lrow[r] = 0.0f; }

    unsigned short* pw = ptile + wave * 16 * LDT;

    for (int kt0 = 0; kt0 < 2048; kt0 += 64) {
        __syncthreads();   // everyone done with previous K/V tiles
#ifdef USE_TDM
        if (wave == 0)     // one wave-level DMA; EXEC-independent, TENSORcnt-tracked
            tdm_load_tile64(kb + (size_t)(bbase + kt0) * 64,
                            (unsigned)(uintptr_t)&ktile[0]);
        for (int i = tid; i < 64 * 64; i += 256) {    // V transpose overlaps the TDM
            const int key = i >> 6, c = i & 63;
            vtile[c * LDT + key] = vb[(size_t)(bbase + kt0 + key) * 64 + c];
        }
        if (wave == 0) __builtin_amdgcn_s_wait_tensorcnt(0);
#else
        for (int i = tid; i < 64 * 64; i += 256) {
            const int key = i >> 6, c = i & 63;
            const size_t g = (size_t)(bbase + kt0 + key) * 64 + c;
            ktile[key * LDT + c] = kb[g];
            vtile[c * LDT + key] = vb[g];
        }
#endif
        __syncthreads();

        // S = (Q K^T) * 0.1 : four 16x16 f32 accs across the 64 keys
        v8f s[4];
#pragma unroll
        for (int n = 0; n < 4; ++n) {
            s[n] = zero8();
            const v16bf bk0 = load_b_fragT(ktile, LDT, n * 16, 0, lane);
            s[n] = wmma_bf16(qa0, bk0, s[n]);
            const v16bf bk1 = load_b_fragT(ktile, LDT, n * 16, 32, lane);
            s[n] = wmma_bf16(qa1, bk1, s[n]);
        }
#pragma unroll
        for (int n = 0; n < 4; ++n)
#pragma unroll
            for (int r = 0; r < 8; ++r) s[n][r] *= 0.1f;   // 1/sqrt(100)

        // online softmax stats: half-wave (16-lane) reductions over C-layout rows
        float alpha[8];
#pragma unroll
        for (int r = 0; r < 8; ++r) {
            float t = fmaxf(fmaxf(s[0][r], s[1][r]), fmaxf(s[2][r], s[3][r]));
#pragma unroll
            for (int d = 1; d < 16; d <<= 1) t = fmaxf(t, __shfl_xor(t, d, 32));
            const float mn = fmaxf(mrow[r], t);
            alpha[r] = __expf(mrow[r] - mn);
            mrow[r] = mn;
        }

        float rsum[8];
#pragma unroll
        for (int r = 0; r < 8; ++r) rsum[r] = 0.0f;
#pragma unroll
        for (int n = 0; n < 4; ++n)
#pragma unroll
            for (int r = 0; r < 8; ++r) {
                const float p = __expf(s[n][r] - mrow[r]);
                rsum[r] += p;
                // C-layout -> LDS row-major P (wave-private; LDS in-order per wave)
                pw[(r + 8 * hh) * LDT + n * 16 + nn] = f2bf(p);
            }
#pragma unroll
        for (int r = 0; r < 8; ++r) {
            float t = rsum[r];
#pragma unroll
            for (int d = 1; d < 16; d <<= 1) t += __shfl_xor(t, d, 32);
            lrow[r] = lrow[r] * alpha[r] + t;
#pragma unroll
            for (int n = 0; n < 4; ++n) o[n][r] *= alpha[r];
        }

        asm volatile("" ::: "memory");   // keep P stores before A-frag reloads

        const v16bf pa0 = load_a_frag(pw, LDT, 0, lane);
        const v16bf pa1 = load_a_frag(pw, LDT, 32, lane);
#pragma unroll
        for (int n = 0; n < 4; ++n) {
            const v16bf bv0 = load_b_fragT(vtile, LDT, n * 16, 0, lane);
            o[n] = wmma_bf16(pa0, bv0, o[n]);
            const v16bf bv1 = load_b_fragT(vtile, LDT, n * 16, 32, lane);
            o[n] = wmma_bf16(pa1, bv1, o[n]);
        }
    }

#pragma unroll
    for (int n = 0; n < 4; ++n)
#pragma unroll
        for (int r = 0; r < 8; ++r) {
            const float v = o[n][r] / lrow[r];
            ob[(size_t)(m0 + r + 8 * hh) * 64 + n * 16 + nn] = f2bf(v);
        }
}

// ---------------------------------------------------------------------------
// Kernel 4: output GEMM.  O[16384,64] bf16 x dr[64,512] bf16 -> out f32
// ---------------------------------------------------------------------------
#define LDD 66

__global__ __launch_bounds__(256) void out_kernel(const unsigned short* __restrict__ vo,
                                                  const unsigned short* __restrict__ dr,
                                                  float* __restrict__ out) {
    __shared__ unsigned short drT[512 * LDD];   // [o][c], ~66KB
    const int tid = threadIdx.x, lane = tid & 31, wave = tid >> 5;
    const int row0 = blockIdx.x * 128;

    for (int i = tid; i < 64 * 512; i += 256) {
        const int c = i >> 9, oo = i & 511;
        drT[oo * LDD + c] = dr[i];
    }
    __syncthreads();

    const int m0 = row0 + wave * 16;
    const v16bf a0 = load_a_frag(vo + (size_t)m0 * 64, 64, 0, lane);
    const v16bf a1 = load_a_frag(vo + (size_t)m0 * 64, 64, 32, lane);
    const int nn = lane & 15, hh = lane >> 4;

    for (int nt = 0; nt < 32; ++nt) {
        v8f acc = zero8();
        const v16bf b0 = load_b_fragT(drT, LDD, nt * 16, 0, lane);
        acc = wmma_bf16(a0, b0, acc);
        const v16bf b1 = load_b_fragT(drT, LDD, nt * 16, 32, lane);
        acc = wmma_bf16(a1, b1, acc);
#pragma unroll
        for (int r = 0; r < 8; ++r)
            out[(size_t)(m0 + r + 8 * hh) * 512 + nt * 16 + nn] = acc[r];
    }
}

// ---------------------------------------------------------------------------
extern "C" void kernel_launch(void* const* d_in, const int* in_sizes, int n_in,
                              void* d_out, int out_size, void* d_ws, size_t ws_size,
                              hipStream_t stream) {
    const float* x     = (const float*)d_in[0];   // [8,2048,512]
    const float* kern  = (const float*)d_in[1];   // [3,512,64]
    const float* dense = (const float*)d_in[2];   // [512,512]
    // d_in[3] = heads (==8), folded into densered_kernel
    float* out = (float*)d_out;                   // [8,2048,512] f32

    const size_t NROW = 16384;                    // B*S
    unsigned short* qb = (unsigned short*)d_ws;
    unsigned short* kb = qb + NROW * 64;
    unsigned short* vb = kb + NROW * 64;
    unsigned short* ob = vb + NROW * 64;
    unsigned short* dr = ob + NROW * 64;          // [64,512] bf16 (total ~8.4MB ws)

    qkv_kernel     <<<128, 256, 0, stream>>>(x, kern, qb, kb, vb);
    densered_kernel<<<128, 256, 0, stream>>>(dense, dr);
    attn_kernel    <<<128, 256, 0, stream>>>(qb, kb, vb, ob);
    out_kernel     <<<128, 256, 0, stream>>>(ob, dr, out);
}
